// MultiHeadAttention_2662879723907
// MI455X (gfx1250) — matrix operand order repaired and checked by reading, not compiled, not run on hardware
//
#include <hip/hip_runtime.h>
#include <stdint.h>

typedef __attribute__((ext_vector_type(16))) __bf16 v16bf;
typedef __attribute__((ext_vector_type(8)))  float  v8f;

union Frag16 {
    v16bf v;
    unsigned short u[16];
    unsigned int   w[8];
};

constexpr int Bc   = 4;
constexpr int Sc   = 1024;
constexpr int HIDc = 1024;
constexpr int Hc   = 16;
constexpr int Dc   = 64;
constexpr int NJ   = 181;   // 2*90+1 relative buckets

// Native conversions: let the backend emit v_cvt_pk_bf16_f32 where available.
__device__ __forceinline__ unsigned short f2bf(float x) {
    __bf16 h = (__bf16)x;
    return __builtin_bit_cast(unsigned short, h);
}
__device__ __forceinline__ unsigned int f2bf2(float lo, float hi_) {
    unsigned int a = f2bf(lo), b = f2bf(hi_);
    return a | (b << 16);
}

// A-fragment K index for 16-bit 16x32 A matrix (contiguous order: k = 16h + i, matching B)
__device__ __forceinline__ int kmapA(int i, int hi) {
    return i | hi;
}

__device__ __forceinline__ float redmax16(float v) {
    for (int off = 8; off; off >>= 1) v = fmaxf(v, __shfl_xor(v, off, 32));
    return v;
}
__device__ __forceinline__ float redsum16(float v) {
    for (int off = 8; off; off >>= 1) v += __shfl_xor(v, off, 32);
    return v;
}

// ---------------------------------------------------------------------------
// Generic C = A @ W^T + bias.  One 16x64 C tile per wave (4 accumulators
// sharing one A fragment), K stepped by 32, bf16 WMMA with fp32 accumulation.
// ---------------------------------------------------------------------------
template <bool ABF16, bool OUTBF16>
__global__ void __launch_bounds__(256)
gemm_wt(const void* __restrict__ Ap, const float* __restrict__ W,
        const float* __restrict__ bias, void* __restrict__ Cp,
        int M, int N, int K) {
    const int lane = threadIdx.x & 31;
    const int wv   = threadIdx.x >> 5;
    const int hi   = lane & 16;
    const int col  = lane & 15;

    const int ntiles = N >> 6;                 // 64-wide N tiles
    int tid = blockIdx.x * (blockDim.x >> 5) + wv;
    int tm  = tid / ntiles;
    int tn  = tid - tm * ntiles;
    if (tm * 16 >= M) return;
    const int m0 = tm * 16, n0 = tn * 64;

    v8f acc[4];
    #pragma unroll
    for (int t = 0; t < 4; ++t) acc[t] = (v8f){0.f,0.f,0.f,0.f,0.f,0.f,0.f,0.f};

    for (int k0 = 0; k0 < K; k0 += 32) {
        Frag16 a;
        if (ABF16) {
            const unsigned short* A = (const unsigned short*)Ap;
            const size_t rowoff = (size_t)(m0 + col) * K + k0;
            #pragma unroll
            for (int j = 0; j < 8; ++j) {
                int kl = kmapA(2 * j, hi);
                a.w[j] = *(const unsigned int*)(A + rowoff + kl);
            }
        } else {
            const float* A = (const float*)Ap;
            const size_t rowoff = (size_t)(m0 + col) * K + k0;
            #pragma unroll
            for (int j = 0; j < 8; ++j) {
                int kl = kmapA(2 * j, hi);
                float2 f2 = *(const float2*)(A + rowoff + kl);
                a.w[j] = f2bf2(f2.x, f2.y);
            }
        }
        #pragma unroll
        for (int t = 0; t < 4; ++t) {
            // B[k][n] = W[n][k]  (row of W is contiguous in k)
            Frag16 bm;
            const float* wrow = W + (size_t)(n0 + t * 16 + col) * K + k0 + hi;
            #pragma unroll
            for (int i = 0; i < 16; i += 4) {
                float4 f4 = *(const float4*)(wrow + i);
                bm.w[i / 2]     = f2bf2(f4.x, f4.y);
                bm.w[i / 2 + 1] = f2bf2(f4.z, f4.w);
            }
            acc[t] = __builtin_amdgcn_wmma_f32_16x16x32_bf16(false, a.v, false, bm.v,
                                                             (short)0, acc[t], false, false);
        }
    }

    #pragma unroll
    for (int t = 0; t < 4; ++t) {
        const int nW = n0 + t * 16 + col;
        const float bb = bias[nW];
        #pragma unroll
        for (int r = 0; r < 8; ++r) {
            int m = m0 + r + (hi >> 1);
            float val = acc[t][r] + bb;
            if (OUTBF16) ((unsigned short*)Cp)[(size_t)m * N + nW] = f2bf(val);
            else         ((float*)Cp)[(size_t)m * N + nW] = val;
        }
    }
}

// ---------------------------------------------------------------------------
// QR[b,q,h,j] = Q[b,q,h,:] . rel_k_emb[j,:]
// Flat GEMM: [B*S*H x 64] @ [64 x 181], fp32 out, N edge-guarded.
// ---------------------------------------------------------------------------
__global__ void __launch_bounds__(256)
qr_kernel(const unsigned short* __restrict__ Qb, const float* __restrict__ relK,
          float* __restrict__ QR) {
    const int lane = threadIdx.x & 31;
    const int wv   = threadIdx.x >> 5;
    const int hi   = lane & 16;
    const int col  = lane & 15;

    const int ntiles = 12;                       // ceil(181/16)
    int tid = blockIdx.x * 8 + wv;
    int tm  = tid / ntiles;
    int tn  = tid - tm * ntiles;
    const int m0 = tm * 16;
    const int n  = tn * 16 + col;
    const bool nok = (n < NJ);

    v8f c = {0.f, 0.f, 0.f, 0.f, 0.f, 0.f, 0.f, 0.f};

    for (int k0 = 0; k0 < Dc; k0 += 32) {
        Frag16 a, bm;
        const size_t rowoff = (size_t)(m0 + col) * Dc + k0;   // Q flat: row*64
        #pragma unroll
        for (int j = 0; j < 8; ++j) {
            int kl = kmapA(2 * j, hi);
            a.w[j] = *(const unsigned int*)(Qb + rowoff + kl);
        }
        if (nok) {
            const float* rrow = relK + (size_t)n * Dc + k0 + hi;
            #pragma unroll
            for (int i = 0; i < 16; i += 4) {
                float4 f4 = *(const float4*)(rrow + i);
                bm.w[i / 2]     = f2bf2(f4.x, f4.y);
                bm.w[i / 2 + 1] = f2bf2(f4.z, f4.w);
            }
        } else {
            #pragma unroll
            for (int j = 0; j < 8; ++j) bm.w[j] = 0u;
        }
        c = __builtin_amdgcn_wmma_f32_16x16x32_bf16(false, a.v, false, bm.v,
                                                    (short)0, c, false, false);
    }
    if (nok) {
        #pragma unroll
        for (int r = 0; r < 8; ++r)
            QR[(size_t)(m0 + r + (hi >> 1)) * NJ + n] = c[r];
    }
}

// ---------------------------------------------------------------------------
// Fused two-pass flash attention + relative-position terms.
// grid (S/64, B*H), 128 threads = 4 waves, 16 q-rows per wave.
// ---------------------------------------------------------------------------
__global__ void __launch_bounds__(128)
attn_kernel(const unsigned short* __restrict__ Qb, const unsigned short* __restrict__ Kb,
            const unsigned short* __restrict__ Vb, const float* __restrict__ QR,
            const float* __restrict__ absE, const int* __restrict__ mask,
            const float* __restrict__ relV, unsigned short* __restrict__ ctx) {
    __shared__ __align__(16) unsigned short sK[32 * 64];
    __shared__ __align__(16) unsigned short sV[32 * 64];
    __shared__ __align__(16) unsigned short sP[4][16 * 32];
    __shared__ float sBand[4][16][184];          // 181 buckets, padded

    const int lane = threadIdx.x & 31;
    const int wv   = threadIdx.x >> 5;
    const int hi   = lane & 16;
    const int hs   = hi >> 1;                    // 0 or 8
    const int col  = lane & 15;
    const int tid  = threadIdx.x;

    const int bh = blockIdx.y;
    const int b  = bh >> 4;
    const int h  = bh & 15;
    const int q0 = blockIdx.x * 64 + wv * 16;

    // Preload the wave's Q tile as two 16x32 bf16 A fragments (d = 0..31, 32..63)
    Frag16 aQ0, aQ1;
    {
        const size_t rowoff = (size_t)(b * Sc + q0 + col) * HIDc + h * Dc;
        #pragma unroll
        for (int j = 0; j < 8; ++j) {
            int kl = kmapA(2 * j, hi);
            aQ0.w[j] = *(const unsigned int*)(Qb + rowoff + kl);
            aQ1.w[j] = *(const unsigned int*)(Qb + rowoff + 32 + kl);
        }
    }

    float rmax[8], rsum[8], rinv[8];
    #pragma unroll
    for (int r = 0; r < 8; ++r) { rmax[r] = -1e30f; rsum[r] = 0.f; }
    v8f O0 = {0.f,0.f,0.f,0.f,0.f,0.f,0.f,0.f};
    v8f O1 = O0, O2 = O0, O3 = O0;

    for (int pass = 0; pass < 2; ++pass) {
        if (pass == 1) {
            #pragma unroll
            for (int r = 0; r < 8; ++r) rinv[r] = 1.0f / rsum[r];
            for (int idx = lane; idx < 16 * 184; idx += 32)
                (&sBand[wv][0][0])[idx] = 0.f;    // per-wave region, no barrier needed
        }
        for (int kt = 0; kt < 32; ++kt) {
            const int k0 = kt * 32;
            __syncthreads();                      // previous tile fully consumed
            // Cooperative staging of K (and V in pass 1) as bf16 tiles [32 x 64]
            for (int c = tid; c < 256; c += 128) {
                int row = c >> 3, seg = c & 7;
                size_t src = (size_t)(b * Sc + k0 + row) * HIDc + h * Dc + seg * 8;
                ((uint4*)sK)[c] = *(const uint4*)(Kb + src);
                if (pass == 1) ((uint4*)sV)[c] = *(const uint4*)(Vb + src);
            }
            __syncthreads();

            #pragma unroll
            for (int ns = 0; ns < 2; ++ns) {
                // B fragments: B[d][n] = K[k0+16*ns+n][d]
                Frag16 bk0, bk1;
                {
                    const unsigned int* p0 =
                        (const unsigned int*)(sK + (ns * 16 + col) * 64 + hi);
                    const unsigned int* p1 =
                        (const unsigned int*)(sK + (ns * 16 + col) * 64 + 32 + hi);
                    #pragma unroll
                    for (int j = 0; j < 8; ++j) { bk0.w[j] = p0[j]; bk1.w[j] = p1[j]; }
                }
                v8f e = {0.f,0.f,0.f,0.f,0.f,0.f,0.f,0.f};
                e = __builtin_amdgcn_wmma_f32_16x16x32_bf16(false, aQ0.v, false, bk0.v,
                                                            (short)0, e, false, false);
                e = __builtin_amdgcn_wmma_f32_16x16x32_bf16(false, aQ1.v, false, bk1.v,
                                                            (short)0, e, false, false);
                #pragma unroll
                for (int r = 0; r < 8; ++r) {
                    const int m  = r + hs;
                    const int qg = q0 + m;
                    const int kg = k0 + ns * 16 + col;
                    float ev = e[r];
                    ev += absE[((size_t)h * Sc + qg) * Sc + kg];
                    int dq = kg - qg;
                    int j  = dq < -90 ? 0 : (dq > 90 ? 180 : dq + 90);
                    ev += QR[((size_t)(b * Sc + qg) * Hc + h) * NJ + j];
                    ev *= 0.125f;                                  // 1/sqrt(64)
                    if (mask[((size_t)(b * Sc + qg)) * Sc + kg] == 0) ev = -1e10f;
                    if (pass == 0) {
                        float tmx = redmax16(ev);
                        float nm  = fmaxf(rmax[r], tmx);
                        float pe  = __expf(ev - nm);
                        float ts  = redsum16(pe);
                        rsum[r] = rsum[r] * __expf(rmax[r] - nm) + ts;
                        rmax[r] = nm;
                    } else {
                        float pr = __expf(ev - rmax[r]) * rinv[r]; // normalized attn
                        atomicAdd(&sBand[wv][m][j], pr);           // ds_add_f32
                        sP[wv][m * 32 + ns * 16 + col] = f2bf(pr);
                    }
                }
            }
            if (pass == 1) {
                // P (C layout) -> A layout via per-wave LDS staging (in-order per wave)
                Frag16 aP;
                #pragma unroll
                for (int j = 0; j < 8; ++j) {
                    int kk = kmapA(2 * j, hi);
                    aP.w[j] = *(const unsigned int*)(&sP[wv][col * 32 + kk]);
                }
                #pragma unroll
                for (int t = 0; t < 4; ++t) {
                    Frag16 bv;
                    #pragma unroll
                    for (int i = 0; i < 16; ++i)
                        bv.u[i] = sV[(i + hi) * 64 + t * 16 + col];
                    v8f* Op = (t == 0) ? &O0 : (t == 1) ? &O1 : (t == 2) ? &O2 : &O3;
                    *Op = __builtin_amdgcn_wmma_f32_16x16x32_bf16(false, aP.v, false, bv.v,
                                                                  (short)0, *Op, false, false);
                }
            }
        }
    }

    // r_x: fold the 181-bucket band through rel_v_emb into the output accumulators
    for (int j = 0; j < NJ; ++j) {
        const float* rv = relV + (size_t)j * Dc;
        float b0 = rv[col], b1 = rv[16 + col], b2 = rv[32 + col], b3 = rv[48 + col];
        #pragma unroll
        for (int r = 0; r < 8; ++r) {
            float bd = sBand[wv][r + hs][j];
            O0[r] += bd * b0; O1[r] += bd * b1; O2[r] += bd * b2; O3[r] += bd * b3;
        }
    }

    // Store context (x + r_x) as bf16, layout [B,S,HID]
    #pragma unroll
    for (int r = 0; r < 8; ++r) {
        const int m = r + hs;
        size_t base = (size_t)(b * Sc + q0 + m) * HIDc + h * Dc + col;
        ctx[base]      = f2bf(O0[r]);
        ctx[base + 16] = f2bf(O1[r]);
        ctx[base + 32] = f2bf(O2[r]);
        ctx[base + 48] = f2bf(O3[r]);
    }
}

// ---------------------------------------------------------------------------
extern "C" void kernel_launch(void* const* d_in, const int* in_sizes, int n_in,
                              void* d_out, int out_size, void* d_ws, size_t ws_size,
                              hipStream_t stream) {
    (void)in_sizes; (void)n_in; (void)out_size; (void)ws_size;

    const float* query = (const float*)d_in[0];
    const float* key   = (const float*)d_in[1];
    const float* value = (const float*)d_in[2];
    const int*   mask  = (const int*)d_in[3];
    const float* absE  = (const float*)d_in[4];
    const float* Wq = (const float*)d_in[5];   const float* bq = (const float*)d_in[6];
    const float* Wk = (const float*)d_in[7];   const float* bk = (const float*)d_in[8];
    const float* Wv = (const float*)d_in[9];   const float* bv = (const float*)d_in[10];
    const float* Wo = (const float*)d_in[11];  const float* bo = (const float*)d_in[12];
    const float* relK = (const float*)d_in[13];
    const float* relV = (const float*)d_in[14];

    char* ws = (char*)d_ws;
    const size_t szBf = (size_t)Bc * Sc * HIDc * sizeof(unsigned short);  // 8.4 MB
    unsigned short* Qb = (unsigned short*)(ws);
    unsigned short* Kb = (unsigned short*)(ws + szBf);
    unsigned short* Vb = (unsigned short*)(ws + 2 * szBf);
    unsigned short* Cb = (unsigned short*)(ws + 3 * szBf);
    float*          QR = (float*)(ws + 4 * szBf);                         // 47.5 MB

    const int M = Bc * Sc, N = HIDc, K = HIDc;
    const int gemmBlocks = (M / 16) * (N / 64) / 8;                       // 512

    gemm_wt<false, true><<<gemmBlocks, 256, 0, stream>>>(query, Wq, bq, Qb, M, N, K);
    gemm_wt<false, true><<<gemmBlocks, 256, 0, stream>>>(key,   Wk, bk, Kb, M, N, K);
    gemm_wt<false, true><<<gemmBlocks, 256, 0, stream>>>(value, Wv, bv, Vb, M, N, K);

    const int qrBlocks = ((Bc * Sc * Hc) / 16) * 12 / 8;                  // 6144
    qr_kernel<<<qrBlocks, 256, 0, stream>>>(Qb, relK, QR);

    attn_kernel<<<dim3(Sc / 64, Bc * Hc), 128, 0, stream>>>(Qb, Kb, Vb, QR, absE,
                                                            mask, relV, Cb);

    gemm_wt<true, false><<<gemmBlocks, 256, 0, stream>>>(Cb, Wo, bo, (float*)d_out,
                                                         M, N, K);
}
